// Attention_40922448396956
// MI455X (gfx1250) — matrix operand order repaired and checked by reading, not compiled, not run on hardware
//
#include <hip/hip_runtime.h>
#include <hip/hip_bf16.h>
#include <stdint.h>

// Problem dims (fixed by reference)
#define BB 4
#define SS 2048
#define DD 256
#define HH 8

typedef __attribute__((ext_vector_type(16))) __bf16         v16bf;
typedef __attribute__((ext_vector_type(16))) unsigned short v16us;
typedef __attribute__((ext_vector_type(8)))  unsigned short v8us;
typedef __attribute__((ext_vector_type(8)))  float          v8f;

__device__ __forceinline__ unsigned short f2bf(float f) {
  return __builtin_bit_cast(unsigned short, __float2bfloat16(f));
}

__device__ __forceinline__ v8f wmma_bf16(v16us a, v16us b, v8f c) {
  return __builtin_amdgcn_wmma_f32_16x16x32_bf16(
      false, __builtin_bit_cast(v16bf, a),
      false, __builtin_bit_cast(v16bf, b),
      (short)0, c, false, false);
}

// ---- CDNA5 async global->LDS copy (ASYNCcnt-tracked) ------------------------
// VDST = VGPR with LDS byte address; VADDR = 64-bit global address.
// Generic addresses of __shared__ data carry the LDS offset in addr[31:0].
__device__ __forceinline__ void async_ld_b128(void* lds_ptr, const void* gptr) {
  uint32_t lds_off = (uint32_t)(uintptr_t)lds_ptr;
  asm volatile("global_load_async_to_lds_b128 %0, %1, off"
               :: "v"(lds_off), "v"((uint64_t)(uintptr_t)gptr)
               : "memory");
}
__device__ __forceinline__ void wait_async0() {
  asm volatile("s_wait_asynccnt 0x0" ::: "memory");
}

// ---- tile loaders -----------------------------------------------------------
// A-matrix 16x32 bf16 layout: lane = half*16+m ; elem e<8 -> K=half*8+e ;
// elem e>=8 -> K=16+half*8+(e-8).
__device__ __forceinline__ v16us a_tile_f32(const float* base, int ld, int row0,
                                            int kbase, int lane) {
  int hf = lane >> 4, m = lane & 15;
  const float* p = base + (size_t)(row0 + m) * ld + kbase + hf * 8;
  v16us o;
#pragma unroll
  for (int e = 0; e < 8; ++e) o[e] = f2bf(p[e]);
#pragma unroll
  for (int e = 0; e < 8; ++e) o[8 + e] = f2bf(p[16 + e]);
  return o;
}

__device__ __forceinline__ v16us a_tile_u16(const unsigned short* base, int ld,
                                            int row0, int kbase, int lane) {
  int hf = lane >> 4, m = lane & 15;
  const unsigned short* p = base + (size_t)(row0 + m) * ld + kbase + hf * 8;
  v8us lo = *(const v8us*)p;
  v8us hi = *(const v8us*)(p + 16);
  return __builtin_shufflevector(lo, hi, 0,1,2,3,4,5,6,7,8,9,10,11,12,13,14,15);
}

// B-matrix 32x16 bf16 layout: lane = khalf*16+n ; elem e<8 -> K=khalf*8+e ;
// elem e>=8 -> K=16+khalf*8+(e-8).
// Source must be stored [n][k] row-major (i.e. transposed vs math B).
__device__ __forceinline__ v16us b_tile_u16(const unsigned short* base, int ld,
                                            int n0, int kbase, int lane) {
  int kh = lane >> 4, n = lane & 15;
  const unsigned short* p = base + (size_t)(n0 + n) * ld + kbase + kh * 8;
  v8us lo = *(const v8us*)p;
  v8us hi = *(const v8us*)(p + 16);
  return __builtin_shufflevector(lo, hi, 0,1,2,3,4,5,6,7,8,9,10,11,12,13,14,15);
}

// ---- kernel 0: Wo [2048,256] f32 -> WoT [256,2048] bf16 ---------------------
__global__ __launch_bounds__(256) void transpose_wo_kernel(
    const float* __restrict__ Wo, unsigned short* __restrict__ WoT) {
  for (int idx = blockIdx.x * 256 + threadIdx.x; idx < 2048 * 256;
       idx += gridDim.x * 256) {
    int k = idx >> 8, n = idx & 255;
    WoT[(size_t)n * 2048 + k] = f2bf(Wo[idx]);
  }
}

// ---- kernel 1: per-head projection GEMM  Y[h] = X @ W[h] + b[h] -------------
// X: [B*S, D] f32.  Out bf16: transposeV=0 -> [b,h,s,d] ; =1 -> [b,h,d,s].
__global__ __launch_bounds__(256) void proj_kernel(
    const float* __restrict__ X, const float* __restrict__ W,
    const float* __restrict__ bias, unsigned short* __restrict__ out,
    int transposeV) {
  __shared__ unsigned short Wl[64 * 264];  // W chunk transposed: [n_local][k]
  const int tid = threadIdx.x, lane = tid & 31, wave = tid >> 5;
  const int h = blockIdx.z;
  const int n0 = blockIdx.y * 64;
  const int row0 = (blockIdx.x * 8 + wave) * 16;

  const float* Wh = W + (size_t)h * DD * DD;
  for (int i = tid; i < 256 * 64; i += 256) {
    int k = i >> 6, nl = i & 63;
    Wl[nl * 264 + k] = f2bf(Wh[(size_t)k * DD + n0 + nl]);
  }
  __syncthreads();

  v8f zero = {0.f, 0.f, 0.f, 0.f, 0.f, 0.f, 0.f, 0.f};
  v8f acc[4] = {zero, zero, zero, zero};

#pragma unroll
  for (int kb = 0; kb < 8; ++kb) {
    v16us a  = a_tile_f32(X, DD, row0, kb * 32, lane);
    v16us b0 = b_tile_u16(Wl, 264,  0, kb * 32, lane);
    v16us b1 = b_tile_u16(Wl, 264, 16, kb * 32, lane);
    v16us b2 = b_tile_u16(Wl, 264, 32, kb * 32, lane);
    v16us b3 = b_tile_u16(Wl, 264, 48, kb * 32, lane);
    acc[0] = wmma_bf16(a, b0, acc[0]);
    acc[1] = wmma_bf16(a, b1, acc[1]);
    acc[2] = wmma_bf16(a, b2, acc[2]);
    acc[3] = wmma_bf16(a, b3, acc[3]);
  }

  const int hf = lane >> 4, nl = lane & 15;
#pragma unroll
  for (int nt = 0; nt < 4; ++nt) {
    int d = n0 + nt * 16 + nl;
    float bv = bias[h * DD + d];
#pragma unroll
    for (int j = 0; j < 8; ++j) {
      int r = row0 + j + 8 * hf;        // global row in [0, B*S)
      int b = r >> 11, s = r & (SS - 1);
      unsigned short val = f2bf(acc[nt][j] + bv);
      if (!transposeV)
        out[(((size_t)(b * HH + h)) * SS + s) * DD + d] = val;
      else
        out[(((size_t)(b * HH + h)) * DD + d) * SS + s] = val;
    }
  }
}

// ---- kernel 2: flash attention, double-buffered async K/V staging -----------
// Qb/Kb: [B*H, S, D] bf16 ; Vt: [B*H, D, S] bf16 ;
// Xo: [B*S, D*H] bf16, interleaved index d*H + h (torch stack(dim=-1) order).
__device__ __forceinline__ void stage_kv_async(
    unsigned short* Kd, unsigned short* Vd,
    const unsigned short* Kh, const unsigned short* Vh, int key0, int tid) {
  // K tile [32 keys x 256 d] : 1024 16B chunks, 4 per thread.
#pragma unroll
  for (int c = 0; c < 4; ++c) {
    int idx = c * 256 + tid;
    int key = idx >> 5, part = idx & 31;
    async_ld_b128(Kd + key * 264 + part * 8,
                  Kh + (size_t)(key0 + key) * DD + part * 8);
  }
  // V tile [256 d x 32 keys] : 1024 16B chunks, 4 per thread.
#pragma unroll
  for (int c = 0; c < 4; ++c) {
    int idx = c * 256 + tid;
    int d = idx >> 2, part = idx & 3;
    async_ld_b128(Vd + d * 40 + part * 8,
                  Vh + (size_t)d * SS + key0 + part * 8);
  }
}

__global__ __launch_bounds__(256) void attn_kernel(
    const unsigned short* __restrict__ Qb, const unsigned short* __restrict__ Kb,
    const unsigned short* __restrict__ Vt, unsigned short* __restrict__ Xo) {
  __shared__ unsigned short Kl[2][32 * 264];  // [key][d] padded (2 x 16.5 KB)
  __shared__ unsigned short Vl[2][256 * 40];  // [d][key] padded (2 x 20 KB)
  __shared__ unsigned short Pl[8 * 512];      // per-wave 16x32 P bounce (8 KB)
  const int tid = threadIdx.x, lane = tid & 31, wave = tid >> 5;
  unsigned short* pbuf = Pl + wave * 512;
  const int bh = blockIdx.y;                  // b*H + h
  const int b = bh >> 3, h = bh & 7;
  const int q0 = (blockIdx.x * 8 + wave) * 16;
  const unsigned short* Qh = Qb + (size_t)bh * SS * DD;
  const unsigned short* Kh = Kb + (size_t)bh * SS * DD;
  const unsigned short* Vh = Vt + (size_t)bh * DD * SS;
  const int hf = lane >> 4, nl = lane & 15;
  const float scale = 0.0625f;                // 1/sqrt(256)

  v16us qa[8];
#pragma unroll
  for (int db = 0; db < 8; ++db) qa[db] = a_tile_u16(Qh, DD, q0, db * 32, lane);

  v8f zero = {0.f, 0.f, 0.f, 0.f, 0.f, 0.f, 0.f, 0.f};
  v8f o[16];
#pragma unroll
  for (int nt = 0; nt < 16; ++nt) o[nt] = zero;
  float mrow[8], lrow[8];
#pragma unroll
  for (int j = 0; j < 8; ++j) { mrow[j] = -__builtin_inff(); lrow[j] = 0.f; }

  // prologue: stage first tile
  stage_kv_async(Kl[0], Vl[0], Kh, Vh, 0, tid);
  wait_async0();
  __syncthreads();

  for (int kb = 0; kb < 64; ++kb) {
    const int cur = kb & 1;
    // prefetch next tile while computing this one (ASYNCcnt in flight)
    if (kb + 1 < 64)
      stage_kv_async(Kl[cur ^ 1], Vl[cur ^ 1], Kh, Vh, (kb + 1) * 32, tid);
    const unsigned short* Kc = Kl[cur];
    const unsigned short* Vc = Vl[cur];

    // scores: c0 = Q . K[key0..key0+15]^T ; c1 = keys +16
    v8f c0 = zero, c1 = zero;
#pragma unroll
    for (int db = 0; db < 8; db += 2) {
      v16us t0 = b_tile_u16(Kc, 264,  0,  db      * 32, lane);
      v16us t1 = b_tile_u16(Kc, 264, 16,  db      * 32, lane);
      v16us t2 = b_tile_u16(Kc, 264,  0, (db + 1) * 32, lane);
      v16us t3 = b_tile_u16(Kc, 264, 16, (db + 1) * 32, lane);
      c0 = wmma_bf16(qa[db],     t0, c0);
      c1 = wmma_bf16(qa[db],     t1, c1);
      c0 = wmma_bf16(qa[db + 1], t2, c0);
      c1 = wmma_bf16(qa[db + 1], t3, c1);
    }
    // online softmax: lane owns rows j + 8*hf, col nl; reduce over 16 lanes
    float alpha[8];
#pragma unroll
    for (int j = 0; j < 8; ++j) {
      float s0 = c0[j] * scale, s1 = c1[j] * scale;
      float mx = fmaxf(s0, s1);
#pragma unroll
      for (int msk = 1; msk < 16; msk <<= 1) mx = fmaxf(mx, __shfl_xor(mx, msk, 32));
      float mnew = fmaxf(mrow[j], mx);
      float p0 = __expf(s0 - mnew), p1 = __expf(s1 - mnew);
      float a  = __expf(mrow[j] - mnew);
      float rs = p0 + p1;
#pragma unroll
      for (int msk = 1; msk < 16; msk <<= 1) rs += __shfl_xor(rs, msk, 32);
      lrow[j] = lrow[j] * a + rs;
      mrow[j] = mnew;
      alpha[j] = a;
      int m = j + 8 * hf;                   // C-layout row -> P row (query)
      pbuf[m * 32 + nl]      = f2bf(p0);    // key-local col  0..15
      pbuf[m * 32 + 16 + nl] = f2bf(p1);    // key-local col 16..31
    }
#pragma unroll
    for (int nt = 0; nt < 16; ++nt)
#pragma unroll
      for (int j = 0; j < 8; ++j) o[nt][j] = o[nt][j] * alpha[j];

    // LDS ops are in-order within a wave: store->load needs no barrier
    v16us pa = a_tile_u16(pbuf, 32, 0, 0, lane);
#pragma unroll
    for (int g = 0; g < 4; ++g) {
      v16us vb0 = b_tile_u16(Vc, 40, (g * 4 + 0) * 16, 0, lane);
      v16us vb1 = b_tile_u16(Vc, 40, (g * 4 + 1) * 16, 0, lane);
      v16us vb2 = b_tile_u16(Vc, 40, (g * 4 + 2) * 16, 0, lane);
      v16us vb3 = b_tile_u16(Vc, 40, (g * 4 + 3) * 16, 0, lane);
      o[g * 4 + 0] = wmma_bf16(pa, vb0, o[g * 4 + 0]);
      o[g * 4 + 1] = wmma_bf16(pa, vb1, o[g * 4 + 1]);
      o[g * 4 + 2] = wmma_bf16(pa, vb2, o[g * 4 + 2]);
      o[g * 4 + 3] = wmma_bf16(pa, vb3, o[g * 4 + 3]);
    }

    wait_async0();        // next tile landed in LDS
    __syncthreads();      // everyone done reading cur, next tile visible
  }

  float rinv[8];
#pragma unroll
  for (int j = 0; j < 8; ++j) rinv[j] = 1.0f / lrow[j];
#pragma unroll
  for (int nt = 0; nt < 16; ++nt) {
    int d = nt * 16 + nl;
#pragma unroll
    for (int j = 0; j < 8; ++j) {
      int s = q0 + j + 8 * hf;
      Xo[((size_t)(b * SS + s)) * (DD * HH) + d * HH + h] = f2bf(o[nt][j] * rinv[j]);
    }
  }
}

// ---- kernel 3: output mix  Y = Xo @ WoT^T + bo ------------------------------
// Xo: [B*S, 2048] bf16 ; WoT: [256, 2048] bf16 ; Y: [B*S, 256] f32.
__global__ __launch_bounds__(256) void outproj_kernel(
    const unsigned short* __restrict__ Xo, const unsigned short* __restrict__ WoT,
    const float* __restrict__ bo, float* __restrict__ Y) {
  __shared__ unsigned short Wl[2][64 * 264];  // panel [n_local][k] (2 x 33 KB)
  const int tid = threadIdx.x, lane = tid & 31, wave = tid >> 5;
  const int n0 = blockIdx.y * 64;
  const int row0 = (blockIdx.x * 8 + wave) * 16;

  v8f zero = {0.f, 0.f, 0.f, 0.f, 0.f, 0.f, 0.f, 0.f};
  v8f acc[4] = {zero, zero, zero, zero};

  // async panel stage: 64 rows x 256 k ushorts = 2048 16B chunks, 8 per thread
  auto stage = [&](unsigned short* dst, int kp) {
#pragma unroll
    for (int c = 0; c < 8; ++c) {
      int idx = c * 256 + tid;
      int nlr = idx >> 5, part = idx & 31;
      async_ld_b128(dst + nlr * 264 + part * 8,
                    WoT + (size_t)(n0 + nlr) * 2048 + kp + part * 8);
    }
  };

  stage(Wl[0], 0);
  wait_async0();
  __syncthreads();

  for (int pp = 0; pp < 8; ++pp) {
    const int cur = pp & 1;
    if (pp + 1 < 8) stage(Wl[cur ^ 1], (pp + 1) * 256);
    const unsigned short* Wc = Wl[cur];
    const int kp = pp * 256;
#pragma unroll
    for (int kb = 0; kb < 8; ++kb) {
      v16us a  = a_tile_u16(Xo, 2048, row0, kp + kb * 32, lane);
      v16us b0 = b_tile_u16(Wc, 264,  0, kb * 32, lane);
      v16us b1 = b_tile_u16(Wc, 264, 16, kb * 32, lane);
      v16us b2 = b_tile_u16(Wc, 264, 32, kb * 32, lane);
      v16us b3 = b_tile_u16(Wc, 264, 48, kb * 32, lane);
      acc[0] = wmma_bf16(a, b0, acc[0]);
      acc[1] = wmma_bf16(a, b1, acc[1]);
      acc[2] = wmma_bf16(a, b2, acc[2]);
      acc[3] = wmma_bf16(a, b3, acc[3]);
    }
    wait_async0();
    __syncthreads();
  }

  const int hf = lane >> 4, nl = lane & 15;
#pragma unroll
  for (int nt = 0; nt < 4; ++nt) {
    int d = n0 + nt * 16 + nl;
    float bv = bo[d];
#pragma unroll
    for (int j = 0; j < 8; ++j) {
      int r = row0 + j + 8 * hf;
      Y[(size_t)r * DD + d] = acc[nt][j] + bv;
    }
  }
}

// ---- host -------------------------------------------------------------------
extern "C" void kernel_launch(void* const* d_in, const int* in_sizes, int n_in,
                              void* d_out, int out_size, void* d_ws, size_t ws_size,
                              hipStream_t stream) {
  const float* k  = (const float*)d_in[0];
  const float* v  = (const float*)d_in[1];
  const float* q  = (const float*)d_in[2];
  const float* Wk = (const float*)d_in[3];
  const float* bk = (const float*)d_in[4];
  const float* Wv = (const float*)d_in[5];
  const float* bv = (const float*)d_in[6];
  const float* Wq = (const float*)d_in[7];
  const float* bq = (const float*)d_in[8];
  const float* Wo = (const float*)d_in[9];
  const float* bo = (const float*)d_in[10];
  float* Y = (float*)d_out;

  char* ws = (char*)d_ws;
  const size_t szT = (size_t)BB * HH * SS * DD * sizeof(unsigned short); // 32 MB
  unsigned short* Kb  = (unsigned short*)(ws);
  unsigned short* Vt  = (unsigned short*)(ws + szT);
  unsigned short* Qb  = (unsigned short*)(ws + 2 * szT);
  unsigned short* Xo  = (unsigned short*)(ws + 3 * szT);
  unsigned short* WoT = (unsigned short*)(ws + 4 * szT);   // 1 MB

  dim3 blk(256);
  transpose_wo_kernel<<<dim3(512), blk, 0, stream>>>(Wo, WoT);
  dim3 gproj(64, 4, 8);                 // 8 waves x 16 rows, 64 cols, per head
  proj_kernel<<<gproj, blk, 0, stream>>>(k, Wk, bk, Kb, 0);
  proj_kernel<<<gproj, blk, 0, stream>>>(v, Wv, bv, Vt, 1);
  proj_kernel<<<gproj, blk, 0, stream>>>(q, Wq, bq, Qb, 0);
  attn_kernel<<<dim3(16, 32), blk, 0, stream>>>(Qb, Kb, Vt, Xo);
  outproj_kernel<<<dim3(64, 4), blk, 0, stream>>>(Xo, WoT, bo, Y);

  (void)in_sizes; (void)n_in; (void)out_size; (void)ws_size;
}
